// GRU_delta_t_75531294867999
// MI455X (gfx1250) — matrix operand order repaired and checked by reading, not compiled, not run on hardware
//
#include <hip/hip_runtime.h>
#include <math.h>

// Problem constants from the reference
#define B_    4096
#define H_    128
#define IN_   16
#define SUB_  4
#define T_    512
#define E_    1024
#define DIN_  65
#define KX    96          // x_in K padded 65 -> 96 (3 x K32 WMMA steps)
#define EVB   64          // events per block
#define NBLK  (E_ / EVB)  // 16 blocks per step

typedef _Float16 h16;
typedef __attribute__((ext_vector_type(16))) _Float16 v16h;
typedef __attribute__((ext_vector_type(8)))  float    v8f;

// ---------------------------------------------------------------------------
// WMMA fragment loaders (CDNA5 wave32 layouts, ISA 7.12.2)
// A 16x32 f16: lane<16 row=lane, elems0..7=K0..7, elems8..15=K16..23
//              lane>=16 row=lane-16, elems0..7=K8..15, elems8..15=K24..31
// ---------------------------------------------------------------------------
__device__ __forceinline__ v16h load_A(const h16* __restrict__ src, int ld) {
  const int lane = threadIdx.x & 31;
  const int row  = lane & 15;
  const int kh   = (lane >> 4) & 1;
  const h16* r = src + row * ld + kh * 8;
  v16h a;
#pragma unroll
  for (int i = 0; i < 8; ++i) a[i] = r[i];
#pragma unroll
  for (int i = 0; i < 8; ++i) a[8 + i] = r[16 + i];
  return a;
}

// B 32x16 f16 built from weight matrix stored row-major as W[O][K]:
// B(k,n) = W[obase+n][kbase+k]. lane<16: col=lane, K=0..7,16..23; lane>=16: K=8..15,24..31.
__device__ __forceinline__ v16h load_B(const h16* __restrict__ w, int ldk) {
  const int lane = threadIdx.x & 31;
  const int col  = lane & 15;
  const int kh   = (lane >> 4) & 1;
  const h16* c = w + col * ldk + kh * 8;
  v16h b;
#pragma unroll
  for (int i = 0; i < 8; ++i) b[i] = c[i];
#pragma unroll
  for (int i = 0; i < 8; ++i) b[8 + i] = c[16 + i];
  return b;
}

__device__ __forceinline__ v8f wmma_f16(v16h a, v16h b, v8f c) {
  return __builtin_amdgcn_wmma_f32_16x16x32_f16(false, a, false, b,
                                                (short)0, c, false, false);
}

// Async gather of 16 bytes from global into LDS (ASYNCcnt-tracked, no VGPRs)
__device__ __forceinline__ void async_g2l_b128(unsigned lds_byte_off,
                                               const void* gaddr) {
  asm volatile("global_load_async_to_lds_b128 %0, %1, off"
               :: "v"(lds_byte_off), "v"(gaddr) : "memory");
}
__device__ __forceinline__ void wait_async0() {
  asm volatile("s_wait_asynccnt 0" ::: "memory");
}

// C/D 16x16 f32 fragment: row m = 8*(lane>=16) + v ; col n = lane&15

// ---------------------------------------------------------------------------
// Prep: zero state + accumulators, convert weights fp32 -> f16 (with padding)
// ---------------------------------------------------------------------------
__global__ void prep_kernel(float* __restrict__ acc, float* __restrict__ hbuf,
                            h16* __restrict__ hb16, float* __restrict__ lt,
                            h16* __restrict__ W1h, h16* __restrict__ W2h,
                            h16* __restrict__ Wihh, h16* __restrict__ Whhh,
                            const float* __restrict__ W1,
                            const float* __restrict__ W2,
                            const float* __restrict__ Wih,
                            const float* __restrict__ Whh) {
  const int tid = blockIdx.x * blockDim.x + threadIdx.x;
  const int nt  = gridDim.x * blockDim.x;
  if (tid < 4) acc[tid] = 0.0f;
  for (int i = tid; i < B_ * H_; i += nt) { hbuf[i] = 0.0f; hb16[i] = (h16)0.0f; }
  for (int i = tid; i < B_; i += nt) lt[i] = 0.0f;
  for (int i = tid; i < H_ * H_; i += nt) W1h[i] = (h16)W1[i];
  for (int i = tid; i < 16 * H_; i += nt) {
    int o = i >> 7, k = i & 127;
    W2h[i] = (o < IN_ - 1) ? (h16)W2[o * H_ + k] : (h16)0.0f;
  }
  for (int i = tid; i < 3 * H_ * KX; i += nt) {
    int o = i / KX, k = i % KX;
    Wihh[i] = (k < DIN_) ? (h16)Wih[o * DIN_ + k] : (h16)0.0f;
  }
  for (int i = tid; i < 3 * H_ * H_; i += nt) Whhh[i] = (h16)Whh[i];
}

// ---------------------------------------------------------------------------
// One timestep: prediction + loss + fused GRU update for 64 events per block
// ---------------------------------------------------------------------------
__global__ void __launch_bounds__(256)
step_kernel(const float* __restrict__ X, const float* __restrict__ M,
            const int* __restrict__ bidx,
            const float* __restrict__ b_ih, const float* __restrict__ b_hh,
            const float* __restrict__ b1, const float* __restrict__ b2,
            const h16* __restrict__ W1h, const h16* __restrict__ W2h,
            const h16* __restrict__ Wihh, const h16* __restrict__ Whhh,
            float* __restrict__ hbuf, h16* __restrict__ hb16,
            float* __restrict__ lt, float* __restrict__ acc, int t) {
  __shared__ int   s_idx[EVB];
  __shared__ h16   s_x[EVB * KX];     // x_in, f16, K padded to 96
  __shared__ h16   s_h[EVB * H_];     // gathered h (f16 mirror, async DMA)
  __shared__ h16   s_a1[EVB * H_];    // relu(h W1^T + b1), f16
  __shared__ float s_xl[EVB * 16];    // loss targets X[:,1:,0]
  __shared__ float s_ml[EVB * 16];    // loss mask M[:,1:]
  __shared__ float s_acc[4];

  const int tid  = threadIdx.x;
  const int wave = tid >> 5;
  const int lane = tid & 31;
  const int col  = lane & 15;
  const int mofs = (lane >> 4) << 3;  // 0 or 8 (C/D fragment row offset)
  const int eb   = blockIdx.x * EVB;
  const float tf = (float)t;

  if (tid < 4) s_acc[tid] = 0.0f;

  // --- phase 0a: indices, time delta, x_in zero padding ---
  if (tid < EVB) {
    int idx = bidx[t * E_ + eb + tid];
    s_idx[tid] = idx;
    float delta = tf - lt[idx];
    s_x[tid * KX + 64] = (h16)delta;
#pragma unroll
    for (int c = 65; c < KX; ++c) s_x[tid * KX + c] = (h16)0.0f;
  }
  __syncthreads();

  // --- phase 0b: async h gather, X tile staging, loss targets, prefetch ---
  {
    // h[idx] gather: f16 rows DMA'd straight into LDS (4 x b128 per thread)
    const unsigned s_h_base = (unsigned)(unsigned long long)(void*)&s_h[0];
#pragma unroll
    for (int k = 0; k < 4; ++k) {
      int chunk = k * 256 + tid;        // 1024 chunks of 16B = 64 rows x 256B
      int e = chunk >> 4;               // event-local row
      int c = chunk & 15;               // 16B chunk within row
      const h16* g = hb16 + (size_t)s_idx[e] * H_ + c * 8;
      async_g2l_b128(s_h_base + (unsigned)(e * 256 + c * 16), (const void*)g);
    }

    const float*  Xt  = X + (size_t)(t * E_ + eb) * (IN_ * SUB_);
    const float4* Xt4 = (const float4*)Xt;
    for (int i = tid; i < EVB * 16; i += 256) {
      int e = i >> 4, c4 = i & 15;
      float4 v = Xt4[e * 16 + c4];
      int base = e * KX + c4 * 4;
      s_x[base + 0] = (h16)v.x;
      s_x[base + 1] = (h16)v.y;
      s_x[base + 2] = (h16)v.z;
      s_x[base + 3] = (h16)v.w;
    }
    const float* Mt = M + (size_t)(t * E_ + eb) * IN_;
    if (tid < EVB) {
#pragma unroll
      for (int ch = 1; ch < IN_; ++ch) {
        s_xl[tid * 16 + ch - 1] = Xt[(size_t)tid * 64 + ch * SUB_];
        s_ml[tid * 16 + ch - 1] = Mt[(size_t)tid * IN_ + ch];
      }
      s_xl[tid * 16 + 15] = 0.0f;
      s_ml[tid * 16 + 15] = 0.0f;
    }

    // Prefetch next step's streaming data (X/M read exactly once -> cold):
    // hide HBM latency of step t+1 behind step t's WMMA work.
    if (t + 1 < T_) {
      const float* Xn = X + (size_t)((t + 1) * E_ + eb) * (IN_ * SUB_);
      __builtin_prefetch(Xn + tid * 16, 0, 1);              // 256 x 64B = 16KB
      if (tid < EVB)
        __builtin_prefetch(M + (size_t)((t + 1) * E_ + eb) * IN_ + tid * 16, 0, 1);
      if (tid == 0)
        __builtin_prefetch(bidx + (t + 1) * E_ + eb, 0, 1);
    }

    wait_async0();   // per-wave: our async LDS writes have landed
  }
  __syncthreads();

  // --- phase 1: a1 = relu(h W1^T + b1)   (wave w owns col-tile jc = w) ---
  {
    const int jc = wave;
    const float bb = b1[jc * 16 + col];
#pragma unroll
    for (int r = 0; r < 4; ++r) {
      v8f c = {};
#pragma unroll
      for (int kb = 0; kb < 4; ++kb)
        c = wmma_f16(load_A(&s_h[(r * 16) * H_ + kb * 32], H_),
                     load_B(&W1h[(jc * 16) * H_ + kb * 32], H_), c);
#pragma unroll
      for (int v = 0; v < 8; ++v) {
        int m = r * 16 + mofs + v;
        float val = c[v] + bb;
        s_a1[m * H_ + jc * 16 + col] = (h16)(val > 0.0f ? val : 0.0f);
      }
    }
  }
  __syncthreads();

  // --- phase 2: p = a1 W2^T + b2, masked loss sums (waves 0..3, t>0) ---
  if (wave < 4 && t > 0) {   // wave-uniform: EXEC all-ones for WMMA
    const int r = wave;
    v8f c = {};
#pragma unroll
    for (int kb = 0; kb < 4; ++kb)
      c = wmma_f16(load_A(&s_a1[(r * 16) * H_ + kb * 32], H_),
                   load_B(&W2h[kb * 32], H_), c);
    if (col < 15) {
      const float bb = b2[col];
      float lmse = 0.0f, lmae = 0.0f, lmape = 0.0f, ltot = 0.0f;
#pragma unroll
      for (int v = 0; v < 8; ++v) {
        int m = r * 16 + mofs + v;
        float p  = c[v] + bb;
        float xl = s_xl[m * 16 + col];
        float ml = s_ml[m * 16 + col];
        float d  = xl - p;
        lmse  += d * d * ml;
        lmae  += fabsf(d) * ml;
        lmape += fabsf(d) / (xl + 1e-8f) * ml;
        ltot  += ml;
      }
      atomicAdd(&s_acc[0], lmse);
      atomicAdd(&s_acc[1], lmae);
      atomicAdd(&s_acc[2], lmape);
      atomicAdd(&s_acc[3], ltot);
    }
  }

  // --- phase 3: fused GRU gates; wave owns 4 of 32 (row-tile, jc) tasks ---
  {
#pragma unroll
    for (int q = 0; q < 4; ++q) {
      const int tk = wave + q * 8;
      const int r  = tk >> 3;
      const int jc = tk & 7;
      const int o  = jc * 16 + col;   // hidden column this lane owns

      const h16* xA = &s_x[(r * 16) * KX];
      const h16* hA = &s_h[(r * 16) * H_];

      v8f cr = {}, cz = {}, cni = {}, cnh = {};
#pragma unroll
      for (int kb = 0; kb < 3; ++kb) {
        v16h a = load_A(xA + kb * 32, KX);
        cr  = wmma_f16(a, load_B(&Wihh[(0 * H_ + jc * 16) * KX + kb * 32], KX), cr);
        cz  = wmma_f16(a, load_B(&Wihh[(1 * H_ + jc * 16) * KX + kb * 32], KX), cz);
        cni = wmma_f16(a, load_B(&Wihh[(2 * H_ + jc * 16) * KX + kb * 32], KX), cni);
      }
#pragma unroll
      for (int kb = 0; kb < 4; ++kb) {
        v16h a = load_A(hA + kb * 32, H_);
        cr  = wmma_f16(a, load_B(&Whhh[(0 * H_ + jc * 16) * H_ + kb * 32], H_), cr);
        cz  = wmma_f16(a, load_B(&Whhh[(1 * H_ + jc * 16) * H_ + kb * 32], H_), cz);
        cnh = wmma_f16(a, load_B(&Whhh[(2 * H_ + jc * 16) * H_ + kb * 32], H_), cnh);
      }
      const float bir = b_ih[o],        bhr = b_hh[o];
      const float biz = b_ih[H_ + o],   bhz = b_hh[H_ + o];
      const float bin = b_ih[2*H_ + o], bhn = b_hh[2*H_ + o];
#pragma unroll
      for (int v = 0; v < 8; ++v) {
        int m = r * 16 + mofs + v;               // event-local row
        size_t grow = (size_t)s_idx[m] * H_ + o; // global h element
        float rg = 1.0f / (1.0f + __expf(-(cr[v] + bir + bhr)));
        float zg = 1.0f / (1.0f + __expf(-(cz[v] + biz + bhz)));
        float ng = tanhf((cni[v] + bin) + rg * (cnh[v] + bhn));
        float ho = hbuf[grow];                   // L2-hot old h (f32)
        float hn = (1.0f - zg) * ng + zg * ho;
        hbuf[grow] = hn;                         // f32 state (recurrence)
        hb16[grow] = (h16)hn;                    // f16 mirror (next gather)
      }
    }
  }
  __syncthreads();

  // --- phase 4: last_t scatter + flush block loss partials ---
  if (tid < EVB) lt[s_idx[tid]] = tf;
  if (tid < 4) atomicAdd(&acc[tid], s_acc[tid]);
}

__global__ void final_kernel(const float* __restrict__ acc,
                             float* __restrict__ out) {
  if (threadIdx.x == 0) {
    float tot = acc[3];
    out[0] = acc[0] / tot;
    out[1] = acc[1] / tot;
    out[2] = acc[2] / tot;
  }
}

// ---------------------------------------------------------------------------
// Workspace layout (bytes):
//   [0,16)        loss accumulators (4 f32), padded to 256
//   +2MB          h state   B*H f32
//   +16KB         last_t    B f32
//   +32KB         W1  f16 128x128
//   +4KB          W2  f16 16x128 (row 15 zero)
//   +72KB         W_ih f16 384x96 (K 65..95 zero)
//   +96KB         W_hh f16 384x128
//   +1MB          h f16 mirror  B*H
// Total ~3.3 MB
// ---------------------------------------------------------------------------
extern "C" void kernel_launch(void* const* d_in, const int* in_sizes, int n_in,
                              void* d_out, int out_size, void* d_ws, size_t ws_size,
                              hipStream_t stream) {
  (void)in_sizes; (void)n_in; (void)out_size; (void)ws_size;
  const float* X    = (const float*)d_in[3];
  const float* M    = (const float*)d_in[4];
  const int*   bidx = (const int*)d_in[5];
  const float* W_ih = (const float*)d_in[7];
  const float* W_hh = (const float*)d_in[8];
  const float* b_ih = (const float*)d_in[9];
  const float* b_hh = (const float*)d_in[10];
  const float* W1   = (const float*)d_in[11];
  const float* b1   = (const float*)d_in[12];
  const float* W2   = (const float*)d_in[13];
  const float* b2   = (const float*)d_in[14];

  char* ws = (char*)d_ws;
  float* acc  = (float*)(ws + 0);
  float* hbuf = (float*)(ws + 256);
  float* lt   = (float*)(ws + 256 + (size_t)B_ * H_ * 4);
  char*  wbase = ws + 256 + (size_t)B_ * H_ * 4 + (size_t)B_ * 4;
  h16* W1h  = (h16*)(wbase);
  h16* W2h  = (h16*)(wbase + 32768);
  h16* Wihh = (h16*)(wbase + 32768 + 4096);
  h16* Whhh = (h16*)(wbase + 32768 + 4096 + 73728);
  h16* hb16 = (h16*)(wbase + 32768 + 4096 + 73728 + 98304);

  prep_kernel<<<512, 256, 0, stream>>>(acc, hbuf, hb16, lt, W1h, W2h, Wihh,
                                       Whhh, W1, W2, W_ih, W_hh);
  for (int t = 0; t < T_; ++t) {
    step_kernel<<<NBLK, 256, 0, stream>>>(X, M, bidx, b_ih, b_hh, b1, b2,
                                          W1h, W2h, Wihh, Whhh, hbuf, hb16,
                                          lt, acc, t);
  }
  final_kernel<<<1, 32, 0, stream>>>(acc, (float*)d_out);
}
